// TransformerBlock_21285857919291
// MI455X (gfx1250) — matrix operand order repaired and checked by reading, not compiled, not run on hardware
//
#include <hip/hip_runtime.h>
#include <math.h>

typedef __bf16 bf16;
typedef __attribute__((ext_vector_type(16))) __bf16 v16bf;
typedef __attribute__((ext_vector_type(8)))  float  v8f;

struct alignas(16) B16x8 { bf16 h[8]; };

__device__ __forceinline__ v8f wmma_bf16f32(v16bf a, v16bf b, v8f c) {
  // D = A(16x32 bf16) * B(32x16 bf16) + C(16x16 f32)
  return __builtin_amdgcn_wmma_f32_16x16x32_bf16(false, a, false, b, (short)0, c,
                                                 false, false);
}

// 16-byte global -> LDS async copy (ASYNCcnt-tracked, no VGPR staging).
__device__ __forceinline__ void async_copy16(void* dst_lds, const void* src_global) {
  unsigned l = (unsigned)(unsigned long long)dst_lds;  // LDS aperture: low 32b = offset
  asm volatile("global_load_async_to_lds_b128 %0, %1, off"
               :: "v"(l), "v"(src_global)
               : "memory");
}
__device__ __forceinline__ void wait_async0() {
  asm volatile("s_wait_asynccnt 0" ::: "memory");
}

// A-fragment (16xK, 16-bit): lane m = lane&15; kb = k0 + (lane>>4)*8
// elements 0..7  <- K kb..kb+7 ; elements 8..15 <- K kb+16..kb+23
__device__ __forceinline__ v16bf frag_A(const bf16* t, int ld, int m0, int k0, int lane) {
  const bf16* p = t + (size_t)(m0 + (lane & 15)) * ld + k0 + ((lane >> 4) << 3);
  B16x8 a0 = *(const B16x8*)p;
  B16x8 a1 = *(const B16x8*)(p + 16);
  v16bf r;
#pragma unroll
  for (int e = 0; e < 8; ++e) { r[e] = a0.h[e]; r[e + 8] = a1.h[e]; }
  return r;
}

// B-fragment from N-major ("Bt": [N][K]) storage: lane n = lane&15; kb = k0+(lane>>4)*8
__device__ __forceinline__ v16bf frag_B(const bf16* t, int ld, int n0, int k0, int lane) {
  const bf16* p = t + (size_t)(n0 + (lane & 15)) * ld + k0 + ((lane >> 4) << 3);
  B16x8 b0 = *(const B16x8*)p;
  B16x8 b1 = *(const B16x8*)(p + 16);
  v16bf r;
#pragma unroll
  for (int e = 0; e < 8; ++e) { r[e] = b0.h[e]; r[e + 8] = b1.h[e]; }
  return r;
}

__global__ void cvt_f32_bf16(const float* __restrict__ in, bf16* __restrict__ out,
                             size_t n) {
  size_t i = (size_t)blockIdx.x * blockDim.x + threadIdx.x;
  if (i < n) out[i] = (bf16)in[i];
}

// C(M,N) = act(A(MxK) * Bt(NxK)^T + bias) [+ resid]; outputs f32 and/or bf16.
// ACT: 0 = identity, 1 = exact GELU.
// Double-buffered: tile k+1 streams into LDS via async copies while tile k computes.
template <int ACT>
__global__ __launch_bounds__(256) void gemm_bf16_kernel(
    const bf16* __restrict__ A, const bf16* __restrict__ Bt,
    const float* __restrict__ bias, const float* __restrict__ resid,
    float* __restrict__ Cf, bf16* __restrict__ Cb, int M, int N, int K) {
  constexpr int BM = 128, BN = 128, BK = 32, LD = 40;  // LD*2B = 80B, 16B aligned
  __shared__ bf16 As[2][BM * LD];
  __shared__ bf16 Bs[2][BN * LD];
  const int tid = threadIdx.x;
  const int lane = tid & 31, wave = tid >> 5;
  const int bm = blockIdx.x * BM, bn = blockIdx.y * BN;
  const int wm = (wave & 3) * 32, wn = (wave >> 2) * 64;  // 4x2 wave grid
  v8f acc[2][4] = {};

  auto stage = [&](int buf, int k0) {
#pragma unroll
    for (int j = 0; j < 2; ++j) {  // 128x32 tiles of A and Bt, 16B chunks
      int c = tid + j * 256;
      int r = c >> 2, kc = (c & 3) << 3;
      async_copy16(&As[buf][r * LD + kc], &A[(size_t)(bm + r) * K + k0 + kc]);
      async_copy16(&Bs[buf][r * LD + kc], &Bt[(size_t)(bn + r) * K + k0 + kc]);
    }
  };

  const int nk = K / BK;
  stage(0, 0);
  for (int kt = 0; kt < nk; ++kt) {
    const int cur = kt & 1;
    wait_async0();       // this wave's async tile loads landed in LDS
    __syncthreads();     // everyone's landed
    if (kt + 1 < nk) stage(cur ^ 1, (kt + 1) * BK);  // overlap with compute below
    v16bf af[2], bfv[4];
#pragma unroll
    for (int i = 0; i < 2; ++i) af[i] = frag_A(As[cur], LD, wm + i * 16, 0, lane);
#pragma unroll
    for (int j = 0; j < 4; ++j) bfv[j] = frag_B(Bs[cur], LD, wn + j * 16, 0, lane);
#pragma unroll
    for (int i = 0; i < 2; ++i)
#pragma unroll
      for (int j = 0; j < 4; ++j) acc[i][j] = wmma_bf16f32(af[i], bfv[j], acc[i][j]);
    __syncthreads();     // all reads of buf `cur` done before it is restaged (kt+2)
  }

  const int hi = lane >> 4, ln = lane & 15;
#pragma unroll
  for (int i = 0; i < 2; ++i)
#pragma unroll
    for (int j = 0; j < 4; ++j) {
      const int n = bn + wn + j * 16 + ln;
#pragma unroll
      for (int r = 0; r < 8; ++r) {
        const int m = bm + wm + i * 16 + hi * 8 + r;
        float v = acc[i][j][r];
        if (bias) v += bias[n];
        if (ACT == 1) v = 0.5f * v * (1.0f + erff(v * 0.70710678118654752f));
        if (resid) v += resid[(size_t)m * N + n];
        const size_t o = (size_t)m * N + n;
        if (Cf) Cf[o] = v;
        if (Cb) Cb[o] = (bf16)v;
      }
    }
}

// Causal flash attention, unscaled scores, NEG=-1e10 mask, bf16 operands, f32 accum.
// Q/K/V stored [b*p][1024] with head h occupying columns 64h..64h+63.
// K tiles stream in async; K/V tiles double-buffered against softmax compute.
__global__ __launch_bounds__(256) void flash_attn_kernel(
    const bf16* __restrict__ Qg, const bf16* __restrict__ Kg,
    const bf16* __restrict__ Vg, float* __restrict__ Z) {
  constexpr int P = 2048, D = 1024, HD = 64, BQ = 128, BKV = 64, LQ = 72, LK = 72;
  __shared__ bf16 Qs[BQ * LQ];          // [q][h]
  __shared__ bf16 Ks[2][BKV * LK];      // [kpos][h]   (Bt layout for S = Q K^T)
  __shared__ bf16 Vt[2][HD * LK];       // [h][kpos]   (Bt layout for O += P V)
  __shared__ bf16 Ps[8 * 16 * LK];      // per-wave P scratch (C-layout -> A-layout)
  const int tid = threadIdx.x, lane = tid & 31, wave = tid >> 5;
  const int qb = blockIdx.x * BQ, head = blockIdx.y, batch = blockIdx.z;
  const size_t bb = (size_t)batch * P;
  const int hoff = head * HD;
#pragma unroll
  for (int j = 0; j < 4; ++j) {  // stage 128x64 Q tile (async)
    int c = tid + j * 256;
    int r = c >> 3, hc = (c & 7) << 3;
    async_copy16(&Qs[r * LQ + hc], &Qg[(bb + qb + r) * D + hoff + hc]);
  }
  auto stageKV = [&](int buf, int k0) {
#pragma unroll
    for (int j = 0; j < 2; ++j) {
      int c = tid + j * 256;
      int r = c >> 3, hc = (c & 7) << 3;
      async_copy16(&Ks[buf][r * LK + hc], &Kg[(bb + k0 + r) * D + hoff + hc]);
      B16x8 t = *(const B16x8*)&Vg[(bb + k0 + r) * D + hoff + hc];
#pragma unroll
      for (int e = 0; e < 8; ++e) Vt[buf][(hc + e) * LK + r] = t.h[e];  // transpose
    }
  };
  const int q0w = wave * 16;
  const int hi = lane >> 4, ln = lane & 15;
  v8f o[4] = {};
  float mrun[8], lrun[8];
#pragma unroll
  for (int r = 0; r < 8; ++r) { mrun[r] = -3.0e38f; lrun[r] = 0.0f; }
  const int nkt = (qb + BQ) / BKV;  // only tiles with k0 <= q_max (causal)
  stageKV(0, 0);
  for (int kt = 0; kt < nkt; ++kt) {
    const int cur = kt & 1;
    const int k0 = kt * BKV;
    wait_async0();      // K (and first-iter Q) async copies landed
    __syncthreads();    // + all waves' Vt ds_stores visible (dscnt waited at barrier)
    if (kt + 1 < nkt) stageKV(cur ^ 1, k0 + BKV);  // overlap with compute below
    v8f s[4] = {};
#pragma unroll
    for (int ks = 0; ks < HD; ks += 32) {  // S(16x64) = q(16x64) . k(64x64)^T
      v16bf aq = frag_A(Qs, LQ, q0w, ks, lane);
#pragma unroll
      for (int j = 0; j < 4; ++j)
        s[j] = wmma_bf16f32(aq, frag_B(Ks[cur], LK, j * 16, ks, lane), s[j]);
    }
#pragma unroll
    for (int r = 0; r < 8; ++r) {  // mask + online softmax per q row
      const int qg = qb + q0w + hi * 8 + r;
#pragma unroll
      for (int j = 0; j < 4; ++j) {
        const int kg = k0 + j * 16 + ln;
        if (kg > qg) s[j][r] = -1.0e10f;
      }
      float vm = fmaxf(fmaxf(s[0][r], s[1][r]), fmaxf(s[2][r], s[3][r]));
#pragma unroll
      for (int off = 1; off < 16; off <<= 1) vm = fmaxf(vm, __shfl_xor(vm, off, 32));
      const float mnew = fmaxf(mrun[r], vm);
      const float sc = expf(mrun[r] - mnew);
      float rs = 0.0f;
#pragma unroll
      for (int j = 0; j < 4; ++j) {
        float pv = expf(s[j][r] - mnew);
        s[j][r] = pv;
        rs += pv;
      }
#pragma unroll
      for (int off = 1; off < 16; off <<= 1) rs += __shfl_xor(rs, off, 32);
      lrun[r] = lrun[r] * sc + rs;
      mrun[r] = mnew;
#pragma unroll
      for (int of = 0; of < 4; ++of) o[of][r] *= sc;
    }
    // C-layout P -> per-wave LDS -> reload as A fragments (bf16)
    bf16* Pw = &Ps[wave * 16 * LK];
#pragma unroll
    for (int j = 0; j < 4; ++j)
#pragma unroll
      for (int r = 0; r < 8; ++r) Pw[(hi * 8 + r) * LK + j * 16 + ln] = (bf16)s[j][r];
    asm volatile("s_wait_dscnt 0" ::: "memory");  // per-wave LDS RAW fence
#pragma unroll
    for (int ks = 0; ks < BKV; ks += 32) {  // O(16x64) += P(16x64) . V(64x64)
      v16bf ap = frag_A(Pw, LK, 0, ks, lane);
#pragma unroll
      for (int of = 0; of < 4; ++of)
        o[of] = wmma_bf16f32(ap, frag_B(Vt[cur], LK, of * 16, ks, lane), o[of]);
    }
    __syncthreads();  // all reads of buf `cur` done before it is restaged (kt+2)
  }
#pragma unroll
  for (int r = 0; r < 8; ++r) {
    const float inv = 1.0f / lrun[r];
    const int qg = qb + q0w + hi * 8 + r;
#pragma unroll
    for (int of = 0; of < 4; ++of)
      Z[(bb + qg) * D + hoff + of * 16 + ln] = o[of][r] * inv;
  }
}

// x1 = LN(x + z): custom LayerNorm — unbiased std (d-1), eps added to std.
__global__ __launch_bounds__(256) void resid_ln_kernel(
    const float* __restrict__ X, const float* __restrict__ Zr,
    const float* __restrict__ wln, const float* __restrict__ bln,
    float* __restrict__ X1, bf16* __restrict__ X1b) {
  constexpr int D = 1024;
  __shared__ float red[8];
  const int tid = threadIdx.x, lane = tid & 31, wv = tid >> 5;
  const size_t base = (size_t)blockIdx.x * D;
  float v[4];
  float s = 0.0f;
#pragma unroll
  for (int i = 0; i < 4; ++i) {
    int c = tid + i * 256;
    v[i] = X[base + c] + Zr[base + c];
    s += v[i];
  }
#pragma unroll
  for (int off = 16; off; off >>= 1) s += __shfl_xor(s, off, 32);
  if (lane == 0) red[wv] = s;
  __syncthreads();
  s = 0.0f;
#pragma unroll
  for (int i = 0; i < 8; ++i) s += red[i];
  const float mean = s * (1.0f / D);
  float sq = 0.0f;
#pragma unroll
  for (int i = 0; i < 4; ++i) {
    v[i] -= mean;
    sq += v[i] * v[i];
  }
  __syncthreads();
#pragma unroll
  for (int off = 16; off; off >>= 1) sq += __shfl_xor(sq, off, 32);
  if (lane == 0) red[wv] = sq;
  __syncthreads();
  sq = 0.0f;
#pragma unroll
  for (int i = 0; i < 8; ++i) sq += red[i];
  const float inv = 1.0f / (sqrtf(sq * (1.0f / (D - 1))) + 1e-4f);
#pragma unroll
  for (int i = 0; i < 4; ++i) {
    int c = tid + i * 256;
    float y = v[i] * inv * wln[c] + bln[c];
    X1[base + c] = y;
    X1b[base + c] = (bf16)y;
  }
}

extern "C" void kernel_launch(void* const* d_in, const int* in_sizes, int n_in,
                              void* d_out, int out_size, void* d_ws, size_t ws_size,
                              hipStream_t stream) {
  (void)in_sizes; (void)n_in; (void)out_size; (void)ws_size;
  const float* x     = (const float*)d_in[0];
  const float* W_K   = (const float*)d_in[1];
  const float* W_Q   = (const float*)d_in[2];
  const float* W_V   = (const float*)d_in[3];
  const float* w_ln  = (const float*)d_in[4];
  const float* b_ln  = (const float*)d_in[5];
  const float* W_in  = (const float*)d_in[6];
  const float* b_in  = (const float*)d_in[7];
  const float* W_out = (const float*)d_in[8];
  const float* b_out = (const float*)d_in[9];

  constexpr size_t BP = 4096, Dd = 1024, MM = 4096;
  char* base = (char*)d_ws;
  size_t off = 0;
  auto carve = [&](size_t bytes) {
    void* p = base + off;
    off += (bytes + 255) & ~(size_t)255;
    return p;
  };
  bf16*  xb   = (bf16*)carve(BP * Dd * 2);
  bf16*  wkb  = (bf16*)carve(Dd * Dd * 2);
  bf16*  wqb  = (bf16*)carve(Dd * Dd * 2);
  bf16*  wvb  = (bf16*)carve(Dd * Dd * 2);
  bf16*  wib  = (bf16*)carve(MM * Dd * 2);
  bf16*  wob  = (bf16*)carve(Dd * MM * 2);
  bf16*  qbuf = (bf16*)carve(BP * Dd * 2);
  bf16*  kbuf = (bf16*)carve(BP * Dd * 2);
  bf16*  vbuf = (bf16*)carve(BP * Dd * 2);
  float* z    = (float*)carve(BP * Dd * 4);
  float* x1   = (float*)carve(BP * Dd * 4);
  bf16*  x1b  = (bf16*)carve(BP * Dd * 2);
  bf16*  hb   = qbuf;  // MLP hidden (32MB) aliases dead q/k/v/z region (40MB)

  dim3 blk(256);
  cvt_f32_bf16<<<dim3((unsigned)(BP * Dd / 256)), blk, 0, stream>>>(x, xb, BP * Dd);
  cvt_f32_bf16<<<dim3((unsigned)(Dd * Dd / 256)), blk, 0, stream>>>(W_K, wkb, Dd * Dd);
  cvt_f32_bf16<<<dim3((unsigned)(Dd * Dd / 256)), blk, 0, stream>>>(W_Q, wqb, Dd * Dd);
  cvt_f32_bf16<<<dim3((unsigned)(Dd * Dd / 256)), blk, 0, stream>>>(W_V, wvb, Dd * Dd);
  cvt_f32_bf16<<<dim3((unsigned)(MM * Dd / 256)), blk, 0, stream>>>(W_in, wib, MM * Dd);
  cvt_f32_bf16<<<dim3((unsigned)(Dd * MM / 256)), blk, 0, stream>>>(W_out, wob, Dd * MM);

  // q/k/v[b*p][i*64+h] = x . W^T
  gemm_bf16_kernel<0><<<dim3(32, 8), blk, 0, stream>>>(xb, wqb, nullptr, nullptr,
                                                       nullptr, qbuf, 4096, 1024, 1024);
  gemm_bf16_kernel<0><<<dim3(32, 8), blk, 0, stream>>>(xb, wkb, nullptr, nullptr,
                                                       nullptr, kbuf, 4096, 1024, 1024);
  gemm_bf16_kernel<0><<<dim3(32, 8), blk, 0, stream>>>(xb, wvb, nullptr, nullptr,
                                                       nullptr, vbuf, 4096, 1024, 1024);
  // z = softmax_causal(q k^T) v  (flash, online softmax)
  flash_attn_kernel<<<dim3(16, 16, 2), blk, 0, stream>>>(qbuf, kbuf, vbuf, z);
  // x1 = LN(x + z)
  resid_ln_kernel<<<dim3(4096), blk, 0, stream>>>(x, z, w_ln, b_ln, x1, x1b);
  // h = gelu(x1 . W_in^T + b_in)
  gemm_bf16_kernel<1><<<dim3(32, 32), blk, 0, stream>>>(x1b, wib, b_in, nullptr,
                                                        nullptr, hb, 4096, 4096, 1024);
  // out = x1 + h . W_out^T + b_out
  gemm_bf16_kernel<0><<<dim3(32, 8), blk, 0, stream>>>(hb, wob, b_out, x1,
                                                       (float*)d_out, nullptr,
                                                       4096, 1024, 4096);
}
